// RowRNN_26860725469729
// MI455X (gfx1250) — matrix operand order repaired and checked by reading, not compiled, not run on hardware
//
#include <hip/hip_runtime.h>

#define DEV __device__ __forceinline__

typedef __attribute__((ext_vector_type(16))) _Float16 v16h;
typedef __attribute__((ext_vector_type(8)))  _Float16 v8h;
typedef __attribute__((ext_vector_type(8)))  float    v8f;
typedef __attribute__((ext_vector_type(4)))  unsigned v4u;
typedef __attribute__((ext_vector_type(8)))  int      v8i;
typedef __attribute__((ext_vector_type(4)))  int      v4i;

#if defined(__has_builtin)
#if __has_builtin(__builtin_amdgcn_tensor_load_to_lds) && __has_builtin(__builtin_amdgcn_s_wait_tensorcnt)
#define HAVE_TDM 1
#endif
#endif

// ---------------- problem dims ----------------
constexpr int B_ = 64, C_ = 3, R_ = 64, COL_ = 64, E_ = 64, H_ = 32, V_ = 258;
constexpr int G_ = 96;              // 3*H (gate width)

// ---------------- workspace layout (units: _Float16 elements) ----------------
// Per-channel GRU weight block, contiguous so ONE TDM transfer stages a channel:
//   [ Wih0 (96x96, K padded 65->96) | Wihr (2x96x32) | Whh (3x96x32) ]
constexpr size_t CHW_WIH0 = 0;
constexpr size_t CHW_WIHR = (size_t)96 * 96;                  // 9216
constexpr size_t CHW_WHH  = CHW_WIHR + (size_t)2 * 96 * 32;   // 15360
constexpr size_t SZ_WCH1  = CHW_WHH + (size_t)3 * 96 * 32;    // 24576 halves = 49152 B
constexpr size_t OFF_WCH  = 0;                                // [C][SZ_WCH1]
constexpr size_t SZ_WCH   = (size_t)C_ * SZ_WCH1;
constexpr size_t OFF_WH2E = OFF_WCH + SZ_WCH;                 // [C][96][32]   N padded 65->96
constexpr size_t SZ_WH2E  = (size_t)C_ * 96 * 32;
constexpr size_t OFF_WAD  = OFF_WH2E + SZ_WH2E;               // [C-1][96][32] N padded 65->96
constexpr size_t SZ_WAD   = (size_t)(C_ - 1) * 96 * 32;
constexpr size_t OFF_WOUT = OFF_WAD + SZ_WAD;                 // [272][32]     N padded 258->272
constexpr size_t SZ_WOUT  = (size_t)272 * 32;
constexpr size_t OFF_COMB = ((OFF_WOUT + SZ_WOUT + 511) / 512) * 512; // [COL][B][96] f16
constexpr size_t SZ_COMB  = (size_t)COL_ * B_ * 96;
constexpr size_t OFF_OUT  = OFF_COMB + SZ_COMB;               // [C][R][COL][B][32] f16 hidden outputs
constexpr size_t SZ_OUT   = (size_t)C_ * R_ * COL_ * B_ * H_;

// ---------------- dynamic LDS layout for gru_chain (bytes) ----------------
// gi (24576) | gh (24576) | h[3] (12288) | weight block (49152) | bih (1152) | bhh (1152)
constexpr size_t LDS_GI   = 0;                                // f32 [64][96]
constexpr size_t LDS_GH   = LDS_GI + (size_t)B_ * G_ * 4;     // f32 [64][96]
constexpr size_t LDS_H    = LDS_GH + (size_t)B_ * G_ * 4;     // f16 [3][64][32]
constexpr size_t LDS_WBLK = LDS_H + (size_t)3 * B_ * H_ * 2;  // f16 [SZ_WCH1]
constexpr size_t LDS_BIH  = LDS_WBLK + SZ_WCH1 * 2;           // f32 [3][96]
constexpr size_t LDS_BHH  = LDS_BIH + (size_t)3 * 96 * 4;     // f32 [3][96]
constexpr size_t LDS_TOTAL = LDS_BHH + (size_t)3 * 96 * 4;    // 112,896 B (< 320 KB WGP LDS)

// ---------------- WMMA fragment helpers (wave32, 16x16x32 f16 -> f32) ----------------
DEV v16h cat16(v8h lo, v8h hi) {
  return __builtin_shufflevector(lo, hi, 0,1,2,3,4,5,6,7,8,9,10,11,12,13,14,15);
}

// A tile 16x32, source row-major (row stride ld halves).
// lane 0-15: M=lane, K = {0..7, 16..23}; lane 16-31: M=lane-16, K = {8..15, 24..31}
DEV v16h load_a(const _Float16* base, int ld, int lane) {
  int m  = lane & 15;
  int kb = (lane >> 4) << 3;
  const _Float16* p = base + m * ld + kb;
  v8h lo = *(const v8h*)(p);
  v8h hi = *(const v8h*)(p + 16);
  return cat16(lo, hi);
}

// B tile 32x16 from a weight block stored [N][K] row-major (= column-major B).
// lane 0-15: N=lane, K = {0..7, 16..23}; lane 16-31: N=lane-16, K = {8..15, 24..31}
DEV v16h load_b(const _Float16* wbase, int ld, int lane) {
  int n  = lane & 15;
  int kb = (lane >> 4) << 3;
  const _Float16* p = wbase + n * ld + kb;
  v8h lo = *(const v8h*)(p);
  v8h hi = *(const v8h*)(p + 16);
  return cat16(lo, hi);
}

// D/C tile 16x16 f32: VGPR v, lane l -> (m = v + 8*(l>>4), n = l&15)
DEV void store_d_f32(float* dst, int ld, v8f d, int lane) {
  int n  = lane & 15;
  int mb = (lane >> 4) << 3;
#pragma unroll
  for (int v = 0; v < 8; v++) dst[(mb + v) * ld + n] = d[v];
}

DEV v8f bcast8(float x) {
  v8f a;
#pragma unroll
  for (int i = 0; i < 8; i++) a[i] = x;
  return a;
}

DEV v8f wmma16(v16h a, v16h b, v8f c) {
  return __builtin_amdgcn_wmma_f32_16x16x32_f16(false, a, false, b, (short)0, c, false, false);
}

DEV float sigmoidf_(float x) { return 1.0f / (1.0f + __expf(-x)); }

// ---- Tensor Data Mover: one 2D (1-row) tile, 8-byte elements, global -> LDS ----
// D# per CDNA5 ISA ch.8: group0 = {count, lds_addr, global_addr, type=2},
// group1 = {data_size=8B, tensor_dim0/1, tile_dim0/1, stride0}. Groups 2/3 unused (<=2D).
// This toolchain's builtin takes 6 operands (g0 v4u, g1 v8i, g2 v4i, g3 v4i, v8i, cpol).
DEV void tdm_load_to_lds(const void* gsrc, void* ldsdst, unsigned nquads) {
#if HAVE_TDM
  unsigned long long ga = (unsigned long long)(size_t)gsrc;
  unsigned lds = (unsigned)(size_t)ldsdst;          // LDS aperture: addr[31:0] = LDS offset
  v4u g0;
  g0[0] = 1u;                                        // count=1 (valid user descriptor)
  g0[1] = lds;                                       // lds_addr (bytes)
  g0[2] = (unsigned)ga;                              // global_addr[31:0]
  g0[3] = ((unsigned)(ga >> 32) & 0x01FFFFFFu) | 0x80000000u; // addr[56:32] | type=2
  v8i g1;
  g1[0] = (int)(3u << 16);                           // data_size = 3 (8 bytes)
  g1[1] = (int)((nquads & 0xFFFFu) << 16);           // tensor_dim0[15:0]
  g1[2] = (int)((nquads >> 16) | (1u << 16));        // tensor_dim0[31:16] | tensor_dim1[15:0]=1
  g1[3] = (int)((nquads & 0xFFFFu) << 16);           // tile_dim0 = nquads
  g1[4] = 1;                                         // tile_dim1 = 1, tile_dim2 = 0
  g1[5] = (int)nquads;                               // tensor_dim0_stride[31:0]
  g1[6] = 0;
  g1[7] = 0;
  v4i gz4 = {0, 0, 0, 0};
  v8i gz8 = {0, 0, 0, 0, 0, 0, 0, 0};
  __builtin_amdgcn_tensor_load_to_lds(g0, g1, gz4, gz4, gz8, 0);
  __builtin_amdgcn_s_wait_tensorcnt(0);
#else
  const v8h* s = (const v8h*)gsrc;
  v8h* d = (v8h*)ldsdst;
  for (unsigned i = 0; i < nquads / 2; i++) d[i] = s[i];
#endif
}

// ================= kernel 1: convert/pad all weights to f16 =================
__global__ void prep_weights(const float* Wih0, const float* Wihr, const float* Whh,
                             const float* Wh2e, const float* Wad,  const float* Wout,
                             _Float16* ws) {
  int tid = blockIdx.x * blockDim.x + threadIdx.x;
  int NT  = gridDim.x * blockDim.x;

  for (size_t i = tid; i < (size_t)C_ * 96 * 96; i += NT) {   // [C][96][65] -> per-ch block
    int c = i / 9216, rem = i % 9216, n = rem / 96, k = rem % 96;
    ws[OFF_WCH + c * SZ_WCH1 + CHW_WIH0 + rem] =
        (k < 65) ? (_Float16)Wih0[((size_t)c * 96 + n) * 65 + k] : (_Float16)0.0f;
  }
  for (size_t i = tid; i < (size_t)C_ * 2 * 96 * 32; i += NT) {
    int c = i / 6144, rem = i % 6144;
    ws[OFF_WCH + c * SZ_WCH1 + CHW_WIHR + rem] = (_Float16)Wihr[i];
  }
  for (size_t i = tid; i < (size_t)C_ * 3 * 96 * 32; i += NT) {
    int c = i / 9216, rem = i % 9216;
    ws[OFF_WCH + c * SZ_WCH1 + CHW_WHH + rem] = (_Float16)Whh[i];
  }
  for (size_t i = tid; i < SZ_WH2E; i += NT) {                // [C][65][32] -> [C][96][32]
    int c = i / 3072, rem = i % 3072, n = rem / 32, k = rem % 32;
    ws[OFF_WH2E + i] = (n < 65) ? (_Float16)Wh2e[((size_t)c * 65 + n) * 32 + k] : (_Float16)0.0f;
  }
  for (size_t i = tid; i < SZ_WAD; i += NT) {                 // [2][65][32] -> [2][96][32]
    int c = i / 3072, rem = i % 3072, n = rem / 32, k = rem % 32;
    ws[OFF_WAD + i] = (n < 65) ? (_Float16)Wad[((size_t)c * 65 + n) * 32 + k] : (_Float16)0.0f;
  }
  for (size_t i = tid; i < SZ_WOUT; i += NT) {                // [258][32] -> [272][32]
    int n = i / 32;
    ws[OFF_WOUT + i] = (n < 258) ? (_Float16)Wout[i] : (_Float16)0.0f;
  }
}

// ================= kernel 2: persistent GRU chain (1 WG, 8 wave32) =================
// ~113 KB dynamic LDS: gate buffers + hidden state + the full per-channel weight
// set (staged once per channel by the Tensor Data Mover), so the latency-critical
// WMMA B operands come from ds_load_b128 instead of L2 round trips.
__global__ __launch_bounds__(256) void gru_chain(const int* __restrict__ x,
                                                 const float* __restrict__ embed,
                                                 const float* __restrict__ bih,
                                                 const float* __restrict__ bhh,
                                                 const float* __restrict__ b_h2e,
                                                 const float* __restrict__ b_ad,
                                                 _Float16* __restrict__ ws) {
  const _Float16* Wh2ep = ws + OFF_WH2E;
  const _Float16* Wadp  = ws + OFF_WAD;
  _Float16* comb   = ws + OFF_COMB;   // [COL][B][96] f16 (L2-hot)
  _Float16* outbuf = ws + OFF_OUT;    // [C][R][COL][B][32] f16

  extern __shared__ char smem[];
  float*    s_gi   = (float*)   (smem + LDS_GI);    // [64][96] gi sums (incl. bih)
  float*    s_gh   = (float*)   (smem + LDS_GH);    // [64][96] gh sums (incl. bhh)
  _Float16* s_h    = (_Float16*)(smem + LDS_H);     // [3][64][32] hidden state
  _Float16* s_wblk = (_Float16*)(smem + LDS_WBLK);  // per-channel weight block
  _Float16* s_wih0 = s_wblk + CHW_WIH0;             // [96][96]
  _Float16* s_wihr = s_wblk + CHW_WIHR;             // [2][96][32]
  _Float16* s_whh  = s_wblk + CHW_WHH;              // [3][96][32]
  float*    s_bih  = (float*)   (smem + LDS_BIH);   // [3][96]
  float*    s_bhh  = (float*)   (smem + LDS_BHH);   // [3][96]

  const int tid  = threadIdx.x;
  const int wave = tid >> 5;
  const int lane = tid & 31;

  for (int c = 0; c < C_; c++) {
    // ---- stage this channel's weights (one TDM transfer) + biases into LDS ----
    if (wave == 0)
      tdm_load_to_lds(ws + OFF_WCH + (size_t)c * SZ_WCH1, s_wblk,
                      (unsigned)(SZ_WCH1 * 2 / 8));           // 6144 x 8B
    for (int i = tid; i < 3 * 96; i += 256) {
      s_bih[i] = bih[c * 3 * 96 + i];
      s_bhh[i] = bhh[c * 3 * 96 + i];
    }
    __syncthreads();

    for (int r = 0; r < R_; r++) {
      const float pos_r = (float)r * (2.0f / 64.0f) - 1.0f;
      const _Float16* prevrow = outbuf + (((size_t)c * R_ + (r - 1)) * COL_) * B_ * H_;
      const _Float16* pchan   = outbuf + (((size_t)(c - 1) * R_ + r) * COL_) * B_ * H_;

      // ---- phase A: comb[col][b][0..95] = feat + prev_h@W_h2e^T + b_h2e (+ p_out@W_ad^T + b_ad)
      // GEMM M = COL*B = 4096 rows (row = col*64+b), N = 96 (65 valid), K = 32 per term.
      for (int t = wave; t < 256 * 6; t += 8) {
        int mt = t / 6, nt = t % 6;
        int n  = nt * 16 + (lane & 15);
        float bias = 0.0f;
        if (n < 65) {
          bias = b_h2e[c * 65 + n];
          if (c > 0) bias += b_ad[(c - 1) * 65 + n];
        }
        v8f acc = bcast8(bias);
        if (r > 0)
          acc = wmma16(load_a(prevrow + (size_t)mt * 16 * 32, 32, lane),
                       load_b(Wh2ep + ((size_t)c * 96 + nt * 16) * 32, 32, lane), acc);
        if (c > 0)
          acc = wmma16(load_a(pchan + (size_t)mt * 16 * 32, 32, lane),
                       load_b(Wadp + ((size_t)(c - 1) * 96 + nt * 16) * 32, 32, lane), acc);
#pragma unroll
        for (int v = 0; v < 8; v++) {
          int m   = mt * 16 + v + ((lane >> 4) << 3);
          int col = m >> 6, b = m & 63;
          float val = acc[v];
          if (n < 64) {
            int xi = x[(((size_t)b * C_ + c) * R_ + r) * COL_ + col];
            val += embed[((size_t)c * V_ + xi) * E_ + n];
          } else if (n == 64) {
            val += pos_r;
          }
          comb[((size_t)col * B_ + b) * 96 + n] = (_Float16)val;   // n>=65 stores exact 0
        }
      }
      __threadfence();
      __syncthreads();

      // zero hidden state at row start (reference: h0 = zeros per row)
      for (int i = tid; i < 3 * B_ * H_; i += 256) s_h[i] = (_Float16)0.0f;
      __syncthreads();

      // ---- phase B: GRU scan over columns ----
      for (int j = 0; j < COL_; j++) {
        if (j + 1 < COL_)
          __builtin_prefetch((const void*)(comb + (size_t)(j + 1) * B_ * 96), 0, 1);

        for (int l = 0; l < 3; l++) {
          // 24 output tiles (4 Mtiles x 6 Ntiles) of [64,96]; 3 tiles per wave
#pragma unroll
          for (int i = 0; i < 3; i++) {
            int t  = wave * 3 + i;
            int mt = t & 3, nt = t >> 2;
            int n  = nt * 16 + (lane & 15);
            // gi = input @ Wih^T + bih
            v8f acc = bcast8(s_bih[l * G_ + n]);
            if (l == 0) {
              const _Float16* Ab = comb + (size_t)j * B_ * 96 + (size_t)mt * 16 * 96;
              const _Float16* Wb = s_wih0 + (size_t)(nt * 16) * 96;
#pragma unroll
              for (int kt = 0; kt < 3; kt++)
                acc = wmma16(load_a(Ab + kt * 32, 96, lane), load_b(Wb + kt * 32, 96, lane), acc);
            } else {
              acc = wmma16(load_a(s_h + (size_t)(l - 1) * B_ * H_ + (size_t)mt * 16 * 32, 32, lane),
                           load_b(s_wihr + (size_t)((l - 1) * 96 + nt * 16) * 32, 32, lane),
                           acc);
            }
            store_d_f32(s_gi + mt * 16 * 96 + nt * 16, 96, acc, lane);
            // gh = h @ Whh^T + bhh   (kept separate: n-gate needs i_n + r*h_n)
            v8f acc2 = bcast8(s_bhh[l * G_ + n]);
            acc2 = wmma16(load_a(s_h + (size_t)l * B_ * H_ + (size_t)mt * 16 * 32, 32, lane),
                          load_b(s_whh + (size_t)(l * 96 + nt * 16) * 32, 32, lane),
                          acc2);
            store_d_f32(s_gh + mt * 16 * 96 + nt * 16, 96, acc2, lane);
          }
          __syncthreads();

          // gate nonlinearities + state update (PyTorch order r,z,n)
          _Float16* outrow = outbuf + (((size_t)c * R_ + r) * COL_ + j) * B_ * H_;
          _Float16* hl = s_h + (size_t)l * B_ * H_;
          for (int idx = tid; idx < B_ * H_; idx += 256) {
            int m = idx >> 5, jj = idx & 31;
            float rr = sigmoidf_(s_gi[m * 96 + jj]      + s_gh[m * 96 + jj]);
            float zz = sigmoidf_(s_gi[m * 96 + 32 + jj] + s_gh[m * 96 + 32 + jj]);
            float nn = tanhf(s_gi[m * 96 + 64 + jj] + rr * s_gh[m * 96 + 64 + jj]);
            float hn = (1.0f - zz) * nn + zz * (float)hl[m * 32 + jj];
            hl[m * 32 + jj] = (_Float16)hn;
            if (l == 2) outrow[m * 32 + jj] = (_Float16)hn;   // top layer -> row output
          }
          __syncthreads();
        }
      }
      __threadfence();   // outbuf row visible to next row's / next channel's phase A
      __syncthreads();
    }
  }
}

// ================= kernel 3: output projection  out = h @ W_out^T + b_out =================
// M = C*R*COL*B = 786432 rows (f16, K=32), N = 258 (padded 272). One Mtile per wave.
__global__ __launch_bounds__(256) void proj_kernel(const _Float16* __restrict__ outbuf,
                                                   const _Float16* __restrict__ Woutp,
                                                   const float* __restrict__ b_out,
                                                   float* __restrict__ out) {
  const int wave = threadIdx.x >> 5;
  const int lane = threadIdx.x & 31;
  const size_t mt = (size_t)blockIdx.x * 8 + wave;          // 49152 Mtiles
  v16h a = load_a(outbuf + mt * 16 * 32, 32, lane);         // A reused across all N tiles
#pragma unroll 1
  for (int nt = 0; nt < 17; nt++) {
    int n = nt * 16 + (lane & 15);
    v8f acc = bcast8((n < 258) ? b_out[n] : 0.0f);
    acc = wmma16(a, load_b(Woutp + (size_t)nt * 16 * 32, 32, lane), acc);
    if (n < 258) {
#pragma unroll
      for (int v = 0; v < 8; v++) {
        size_t m = mt * 16 + v + ((lane >> 4) << 3);        // row = ((c*R+r)*COL+col)*B + b
        size_t b = m & 63, col = (m >> 6) & 63, r = (m >> 12) & 63, c = m >> 18;
        out[(((b * C_ + c) * R_ + r) * COL_ + col) * 258 + n] = acc[v];
      }
    }
  }
}

// ================= host entry =================
extern "C" void kernel_launch(void* const* d_in, const int* in_sizes, int n_in,
                              void* d_out, int out_size, void* d_ws, size_t ws_size,
                              hipStream_t stream) {
  const int*   x     = (const int*)d_in[0];
  const float* embed = (const float*)d_in[1];
  const float* Wih0  = (const float*)d_in[2];
  const float* Wihr  = (const float*)d_in[3];
  const float* Whh   = (const float*)d_in[4];
  const float* bih   = (const float*)d_in[5];
  const float* bhh   = (const float*)d_in[6];
  const float* Wh2e  = (const float*)d_in[7];
  const float* bh2e  = (const float*)d_in[8];
  const float* Wad   = (const float*)d_in[9];
  const float* bad   = (const float*)d_in[10];
  const float* Wout  = (const float*)d_in[11];
  const float* bout  = (const float*)d_in[12];
  _Float16* ws  = (_Float16*)d_ws;
  float*    out = (float*)d_out;

  prep_weights<<<96, 256, 0, stream>>>(Wih0, Wihr, Whh, Wh2e, Wad, Wout, ws);
  gru_chain<<<1, 256, (unsigned)LDS_TOTAL, stream>>>(x, embed, bih, bhh, bh2e, bad, ws);
  proj_kernel<<<6144, 256, 0, stream>>>(ws + OFF_OUT, ws + OFF_WOUT, bout, out);
}
